// DetSegTransformerDecoder_34076270527050
// MI455X (gfx1250) — matrix operand order repaired and checked by reading, not compiled, not run on hardware
//
#include <hip/hip_runtime.h>
#include <hip/hip_bf16.h>
#include <math.h>

typedef __attribute__((ext_vector_type(16))) _Float16 v16h;
typedef __attribute__((ext_vector_type(8)))  float    v8f;

#define DD   128
#define PP   4
#define NLVL 4
#define NCAM 6
#define HBEV 200
#define WBEV 200
#define QNUM (HBEV*WBEV)
#define IMGH 256
#define IMGW 704
#define EPSF 1e-5f

// ---------------------------------------------------------------------------
// elementwise helpers
// ---------------------------------------------------------------------------
__global__ void k_copy(const float* __restrict__ in, float* __restrict__ out, int n) {
  int i = blockIdx.x * blockDim.x + threadIdx.x;
  if (i < n) out[i] = in[i];
}

__global__ void k_add(const float* __restrict__ a, const float* __restrict__ b,
                      float* __restrict__ out, int n) {
  int i = blockIdx.x * blockDim.x + threadIdx.x;
  if (i < n) out[i] = a[i] + b[i];
}

// (NC, D, H, W) -> (NC, H, W, D)
__global__ void k_transpose_feat(const float* __restrict__ in, float* __restrict__ out,
                                 int H, int W) {
  int idx = blockIdx.x * blockDim.x + threadIdx.x;
  int tot = NCAM * DD * H * W;
  if (idx >= tot) return;
  int c = idx % DD;
  int r = idx / DD;
  int x = r % W; r /= W;
  int y = r % H; r /= H;
  int n = r;
  out[idx] = in[(((size_t)n * DD + c) * H + y) * W + x];
}

// ---------------------------------------------------------------------------
// fragment packing.
// A-fragment (16x32 f16): lane L holds row m = L&15; halves e map to
//   k = (L>=16 ? 8 : 0) + (e<8 ? e : e+8)
// B-fragment (32x16 f16): lane L holds col n = L&15; halves e map to
//   k = (L>=16 ? 8 : 0) + (e<8 ? e : e+8)
// Each (tile, lane) record is 16 contiguous halves (32B) so the GEMM hot loop
// issues a single 32B vector load per operand per WMMA.
// tile order: A -> mt*ktiles + kt ; B -> kt*ntiles + nt
// ---------------------------------------------------------------------------
__global__ void k_pack_a(const float* __restrict__ A, _Float16* __restrict__ out,
                         int M, int K) {
  int ktiles = K >> 5;
  int tiles  = (M >> 4) * ktiles;
  int wv = blockIdx.x * 8 + (threadIdx.x >> 5);
  if (wv >= tiles) return;
  int mt = wv / ktiles;
  int kt = wv % ktiles;
  int lane = threadIdx.x & 31;
  int m   = (mt << 4) + (lane & 15);
  int kba = (lane >> 4) << 3;
  const float* src = A + (size_t)m * K + (kt << 5) + kba;
  const float4 a0 = *(const float4*)(src + 0);
  const float4 a1 = *(const float4*)(src + 4);
  const float4 a2 = *(const float4*)(src + 16);
  const float4 a3 = *(const float4*)(src + 20);
  v16h af;
  af[0]  = (_Float16)a0.x; af[1]  = (_Float16)a0.y; af[2]  = (_Float16)a0.z; af[3]  = (_Float16)a0.w;
  af[4]  = (_Float16)a1.x; af[5]  = (_Float16)a1.y; af[6]  = (_Float16)a1.z; af[7]  = (_Float16)a1.w;
  af[8]  = (_Float16)a2.x; af[9]  = (_Float16)a2.y; af[10] = (_Float16)a2.z; af[11] = (_Float16)a2.w;
  af[12] = (_Float16)a3.x; af[13] = (_Float16)a3.y; af[14] = (_Float16)a3.z; af[15] = (_Float16)a3.w;
  *(v16h*)(out + ((size_t)wv * 32 + lane) * 16) = af;
}

__global__ void k_pack_b(const float* __restrict__ W, _Float16* __restrict__ out,
                         int K, int N) {
  int ntiles = N >> 4;
  int tiles  = (K >> 5) * ntiles;
  int wv = blockIdx.x * 8 + (threadIdx.x >> 5);
  if (wv >= tiles) return;
  int kt = wv / ntiles;
  int nt = wv % ntiles;
  int lane = threadIdx.x & 31;
  int n   = (nt << 4) + (lane & 15);
  int kbb = (lane >> 4) << 3;
  const float* src = W + (size_t)((kt << 5) + kbb) * N + n;
  v16h bf;
#pragma unroll
  for (int e = 0; e < 16; ++e) bf[e] = (_Float16)src[(size_t)(e < 8 ? e : e + 8) * N];
  *(v16h*)(out + ((size_t)wv * 32 + lane) * 16) = bf;
}

// ---------------------------------------------------------------------------
// position embedding: relu(pos @ w1 + b1) @ w2 + b2   (3 -> 256 -> 128)
// ---------------------------------------------------------------------------
__global__ void k_pos_embed(const float* __restrict__ bev_pos,
                            const float* __restrict__ w1, const float* __restrict__ b1,
                            const float* __restrict__ w2, const float* __restrict__ b2,
                            float* __restrict__ out) {
  __shared__ float hid[256];
  int q0 = blockIdx.x;
  int t  = threadIdx.x;
  float p0 = bev_pos[q0 * 3 + 0];
  float p1 = bev_pos[q0 * 3 + 1];
  float p2 = bev_pos[q0 * 3 + 2];
  float h = p0 * w1[0 * 256 + t] + p1 * w1[1 * 256 + t] + p2 * w1[2 * 256 + t] + b1[t];
  hid[t] = fmaxf(h, 0.f);
  __syncthreads();
  if (t < DD) {
    float s = b2[t];
    for (int k = 0; k < 256; ++k) s += hid[k] * w2[k * DD + t];
    out[(size_t)q0 * DD + t] = s;
  }
}

// ---------------------------------------------------------------------------
// packed WMMA GEMM: out[M,N] = act(A @ W + bias)
// A, B pre-packed into per-lane fragment records (32B each).
// Register blocking: each wave computes a 16x64 C strip (4 accumulators);
// one A-fragment load feeds 4 WMMAs.
// act: 0 = none, 1 = relu, 2 = gelu(exact)
// ---------------------------------------------------------------------------
__global__ void k_gemm_pk(const _Float16* __restrict__ Ap, const _Float16* __restrict__ Bp,
                          const float* __restrict__ bias, float* __restrict__ out,
                          int M, int K, int N, int act) {
  int mtiles  = M >> 4;
  int ntiles  = N >> 4;
  int ktiles  = K >> 5;
  int ngroups = ntiles >> 2;          // 4 N-tiles per wave
  int wv = blockIdx.x * 4 + (threadIdx.x >> 5);
  if (wv >= mtiles * ngroups) return;
  int mt  = wv / ngroups;
  int ng  = wv % ngroups;
  int nt0 = ng << 2;
  int lane = threadIdx.x & 31;

  const v16h* Aptr = (const v16h*)Ap + (size_t)mt * ktiles * 32 + lane;
  const v16h* Bptr = (const v16h*)Bp + (size_t)nt0 * 32 + lane;

  v8f acc[4] = {{}, {}, {}, {}};
  for (int kt = 0; kt < ktiles; ++kt) {
    __builtin_prefetch(Aptr + (size_t)(kt + 1) * 32, 0, 1);   // global_prefetch_b8
    v16h a = Aptr[(size_t)kt * 32];
    const v16h* Bk = Bptr + (size_t)kt * ntiles * 32;
#pragma unroll
    for (int j = 0; j < 4; ++j) {
      v16h b = Bk[(size_t)j * 32];
      acc[j] = __builtin_amdgcn_wmma_f32_16x16x32_f16(false, a, false, b,
                                                      (short)0, acc[j], false, false);
    }
  }

  int mb = (mt << 4) + ((lane >> 4) << 3);
#pragma unroll
  for (int j = 0; j < 4; ++j) {
    int nc_ = ((nt0 + j) << 4) + (lane & 15);
    float bv = bias[nc_];
#pragma unroll
    for (int r = 0; r < 8; ++r) {
      float v = acc[j][r] + bv;
      if (act == 1)      v = fmaxf(v, 0.f);
      else if (act == 2) v = 0.5f * v * (1.f + erff(v * 0.70710678118654752f));
      out[(size_t)(mb + r) * N + nc_] = v;
    }
  }
}

// ---------------------------------------------------------------------------
// 5x5 SAME conv on (HBEV,WBEV,128) as implicit WMMA GEMM, K = 25*128 = 3200.
// B (weights) pre-packed. Each wave computes a full 16x128 output strip
// (8 accumulators), so the expensive f32 halo gather + f16 convert of the
// A-fragment happens once per k-chunk and feeds 8 WMMAs.
// ---------------------------------------------------------------------------
__global__ void k_conv5x5_wmma(const float* __restrict__ h, const _Float16* __restrict__ Bp,
                               const float* __restrict__ bias, float* __restrict__ out) {
  const int ntiles = DD >> 4;                 // 8
  int wv = blockIdx.x * 4 + (threadIdx.x >> 5);
  if (wv >= (QNUM >> 4)) return;
  int m0 = wv << 4;

  int lane = threadIdx.x & 31;
  int kba  = (lane >> 4) << 3;
  int m    = m0 + (lane & 15);
  int oy = m / WBEV;
  int ox = m % WBEV;

  const v16h* Bbase = (const v16h*)Bp + lane;

  v8f acc[8] = {{}, {}, {}, {}, {}, {}, {}, {}};
  for (int tap = 0; tap < 25; ++tap) {
    int iy = oy + tap / 5 - 2;
    int ix = ox + tap % 5 - 2;
    bool valid = (iy >= 0) & (iy < HBEV) & (ix >= 0) & (ix < WBEV);
    const float* hrow = h + ((size_t)(valid ? iy : 0) * WBEV + (valid ? ix : 0)) * DD;
    for (int kc = 0; kc < DD; kc += 32) {
      v16h af;
      if (valid) {
        const float4 a0 = *(const float4*)(hrow + kc + kba + 0);
        const float4 a1 = *(const float4*)(hrow + kc + kba + 4);
        const float4 a2 = *(const float4*)(hrow + kc + kba + 16);
        const float4 a3 = *(const float4*)(hrow + kc + kba + 20);
        af[0]  = (_Float16)a0.x; af[1]  = (_Float16)a0.y; af[2]  = (_Float16)a0.z; af[3]  = (_Float16)a0.w;
        af[4]  = (_Float16)a1.x; af[5]  = (_Float16)a1.y; af[6]  = (_Float16)a1.z; af[7]  = (_Float16)a1.w;
        af[8]  = (_Float16)a2.x; af[9]  = (_Float16)a2.y; af[10] = (_Float16)a2.z; af[11] = (_Float16)a2.w;
        af[12] = (_Float16)a3.x; af[13] = (_Float16)a3.y; af[14] = (_Float16)a3.z; af[15] = (_Float16)a3.w;
      } else {
#pragma unroll
        for (int e = 0; e < 16; ++e) af[e] = (_Float16)0.f;
      }
      int ktg = (tap * DD + kc) >> 5;           // global k-tile index
      const v16h* Bk = Bbase + (size_t)ktg * ntiles * 32;
#pragma unroll
      for (int j = 0; j < 8; ++j) {
        v16h bf = Bk[(size_t)j * 32];
        acc[j] = __builtin_amdgcn_wmma_f32_16x16x32_f16(false, af, false, bf,
                                                        (short)0, acc[j], false, false);
      }
    }
  }

  int mb = m0 + ((lane >> 4) << 3);
#pragma unroll
  for (int j = 0; j < 8; ++j) {
    int nc_ = (j << 4) + (lane & 15);
    float bv = bias[nc_];
#pragma unroll
    for (int r = 0; r < 8; ++r)
      out[(size_t)(mb + r) * DD + nc_] = acc[j][r] + bv;
  }
}

// ---------------------------------------------------------------------------
// residual add + LayerNorm over D=128, in place on q.  block = 128 threads
// ---------------------------------------------------------------------------
__global__ void k_addnorm(float* __restrict__ q, const float* __restrict__ h,
                          const float* __restrict__ g, const float* __restrict__ b) {
  __shared__ float red[8];
  int q0 = blockIdx.x;
  int t  = threadIdx.x;
  float x = q[(size_t)q0 * DD + t] + h[(size_t)q0 * DD + t];
  float s = x, s2 = x * x;
#pragma unroll
  for (int o = 16; o > 0; o >>= 1) {
    s  += __shfl_xor(s,  o, 32);
    s2 += __shfl_xor(s2, o, 32);
  }
  if ((t & 31) == 0) { red[t >> 5] = s; red[4 + (t >> 5)] = s2; }
  __syncthreads();
  float ts  = red[0] + red[1] + red[2] + red[3];
  float ts2 = red[4] + red[5] + red[6] + red[7];
  float m   = ts * (1.f / DD);
  float var = ts2 * (1.f / DD) - m * m;
  q[(size_t)q0 * DD + t] = (x - m) * rsqrtf(var + 1e-5f) * g[t] + b[t];
}

// ---------------------------------------------------------------------------
// offsets (q @ off_w + off_b) and softmaxed scale weights per query
// ---------------------------------------------------------------------------
__global__ void k_offsw(const float* __restrict__ q,
                        const float* __restrict__ off_w, const float* __restrict__ off_b,
                        const float* __restrict__ sw_w,  const float* __restrict__ sw_b,
                        float* __restrict__ off_out, float* __restrict__ sw_out) {
  __shared__ float qs[DD];
  __shared__ float sl[16];
  int q0 = blockIdx.x;
  int t  = threadIdx.x;
  qs[t] = q[(size_t)q0 * DD + t];
  __syncthreads();
  if (t < 12) {
    float s = off_b[t];
    for (int c = 0; c < DD; ++c) s += qs[c] * off_w[c * 12 + t];
    off_out[(size_t)q0 * 12 + t] = s;
  } else if (t >= 32 && t < 48) {
    int j = t - 32;
    float s = sw_b[j];
    for (int c = 0; c < DD; ++c) s += qs[c] * sw_w[c * 16 + j];
    sl[j] = s;
  }
  __syncthreads();
  if (t < PP) {
    float mx = sl[t * 4];
    for (int l = 1; l < NLVL; ++l) mx = fmaxf(mx, sl[t * 4 + l]);
    float e[NLVL], sum = 0.f;
    for (int l = 0; l < NLVL; ++l) { e[l] = __expf(sl[t * 4 + l] - mx); sum += e[l]; }
    float inv = 1.f / sum;
    for (int l = 0; l < NLVL; ++l) sw_out[(size_t)q0 * 16 + t * 4 + l] = e[l] * inv;
  }
}

// ---------------------------------------------------------------------------
// multi-camera multi-level deformable sampling. block = query, thread = channel
// featT: per level (NC, Hl, Wl, D) contiguous, level base offsets baked in
// ---------------------------------------------------------------------------
__global__ void k_sample(const float* __restrict__ bev_pos,
                         const float* __restrict__ off, const float* __restrict__ sw,
                         const float* __restrict__ l2i, const float* __restrict__ featT,
                         float* __restrict__ acc) {
  const int LH[NLVL]   = {32, 16, 8, 4};
  const int LW[NLVL]   = {88, 44, 22, 11};
  const int LOFF[NLVL] = {0, 2162688, 2703360, 2838528};

  int q0 = blockIdx.x;
  int c  = threadIdx.x;

  float rx = bev_pos[q0 * 3 + 0] * 100.f - 50.f;
  float ry = bev_pos[q0 * 3 + 1] * 100.f - 50.f;
  float rz = bev_pos[q0 * 3 + 2] * 8.f   - 5.f;

  float ap[PP] = {0.f, 0.f, 0.f, 0.f};

  for (int n = 0; n < NCAM; ++n) {
    const float* M = l2i + n * 16;
#pragma unroll
    for (int p = 0; p < PP; ++p) {
      float x = rx + off[(size_t)q0 * 12 + p * 3 + 0];
      float y = ry + off[(size_t)q0 * 12 + p * 3 + 1];
      float z = rz + off[(size_t)q0 * 12 + p * 3 + 2];
      float u0 = M[0] * x + M[1] * y + M[2]  * z + M[3];
      float v0 = M[4] * x + M[5] * y + M[6]  * z + M[7];
      float w0 = M[8] * x + M[9] * y + M[10] * z + M[11];
      float zc = fmaxf(w0, EPSF);
      float u = u0 / (zc * IMGW);
      float v = v0 / (zc * IMGH);
      if (w0 > EPSF && u >= 0.f && u <= 1.f && v >= 0.f && v <= 1.f) {
#pragma unroll
        for (int l = 0; l < NLVL; ++l) {
          int Hl = LH[l], Wl = LW[l];
          float fx = u * Wl - 0.5f;
          float fy = v * Hl - 0.5f;
          int x0 = (int)floorf(fx);
          int y0 = (int)floorf(fy);
          float wx = fx - (float)x0;
          float wy = fy - (float)y0;
          const float* base = featT + LOFF[l] + (size_t)n * Hl * Wl * DD;
          float s = 0.f;
#pragma unroll
          for (int ty = 0; ty < 2; ++ty) {
#pragma unroll
            for (int tx = 0; tx < 2; ++tx) {
              int yy = y0 + ty, xx = x0 + tx;
              if (yy >= 0 && yy < Hl && xx >= 0 && xx < Wl) {
                float wt = (tx ? wx : 1.f - wx) * (ty ? wy : 1.f - wy);
                s += base[((size_t)yy * Wl + xx) * DD + c] * wt;
              }
            }
          }
          ap[p] += s * sw[(size_t)q0 * 16 + p * 4 + l];
        }
      }
    }
  }
#pragma unroll
  for (int p = 0; p < PP; ++p)
    acc[(size_t)q0 * (PP * DD) + p * DD + c] = ap[p];
}

// ---------------------------------------------------------------------------
// host driver
// ---------------------------------------------------------------------------
extern "C" void kernel_launch(void* const* d_in, const int* in_sizes, int n_in,
                              void* d_out, int out_size, void* d_ws, size_t ws_size,
                              hipStream_t stream) {
  (void)in_sizes; (void)n_in; (void)out_size; (void)ws_size;

  const float* feat[4]   = {(const float*)d_in[0], (const float*)d_in[1],
                            (const float*)d_in[2], (const float*)d_in[3]};
  const float* l2i       = (const float*)d_in[4];
  const float* bev_query = (const float*)d_in[5];
  const float* bev_pos   = (const float*)d_in[6];
  const float* pe_w1 = (const float*)d_in[7];   const float* pe_b1 = (const float*)d_in[8];
  const float* pe_w2 = (const float*)d_in[9];   const float* pe_b2 = (const float*)d_in[10];
  const float* conv1_w = (const float*)d_in[11]; const float* conv1_b = (const float*)d_in[12];
  const float* conv2_w = (const float*)d_in[13]; const float* conv2_b = (const float*)d_in[14];
  const float* off_w = (const float*)d_in[15];  const float* off_b = (const float*)d_in[16];
  const float* sw_w  = (const float*)d_in[17];  const float* sw_b  = (const float*)d_in[18];
  const float* cp_w1 = (const float*)d_in[19];  const float* cp_b1 = (const float*)d_in[20];
  const float* cp_w2 = (const float*)d_in[21];  const float* cp_b2 = (const float*)d_in[22];
  const float* cp_w3 = (const float*)d_in[23];  const float* cp_b3 = (const float*)d_in[24];
  const float* ffn_w1 = (const float*)d_in[25]; const float* ffn_b1 = (const float*)d_in[26];
  const float* ffn_w2 = (const float*)d_in[27]; const float* ffn_b2 = (const float*)d_in[28];
  const float* n1_g = (const float*)d_in[29];   const float* n1_b = (const float*)d_in[30];
  const float* n2_g = (const float*)d_in[31];   const float* n2_b = (const float*)d_in[32];
  const float* n3_g = (const float*)d_in[33];   const float* n3_b = (const float*)d_in[34];

  float* q = (float*)d_out;   // persistent query buffer == final output

  // workspace carve
  char* ws = (char*)d_ws;
  size_t off = 0;
  auto carve = [&](size_t bytes) -> void* {
    void* p = ws + off;
    off = (off + bytes + 255) & ~(size_t)255;
    return p;
  };
  float* qe    = (float*)carve((size_t)QNUM * DD * 4);
  float* hbuf  = (float*)carve((size_t)QNUM * DD * 4);
  float* hout  = (float*)carve((size_t)QNUM * DD * 4);
  float* pe    = (float*)carve((size_t)QNUM * DD * 4);
  float* offb  = (float*)carve((size_t)QNUM * 12 * 4);
  float* swb   = (float*)carve((size_t)QNUM * 16 * 4);
  float* accb  = (float*)carve((size_t)QNUM * (PP * DD) * 4);
  float* h1b   = (float*)carve((size_t)QNUM * (PP * DD) * 4);
  float* featT = (float*)carve((size_t)2872320 * 4);
  _Float16* apack  = (_Float16*)carve((size_t)QNUM * 512 * 2);   // packed-A scratch (max K=512)
  _Float16* w_conv1 = (_Float16*)carve((size_t)128 * 128 * 2);
  _Float16* w_conv2 = (_Float16*)carve((size_t)3200 * 128 * 2);
  _Float16* w_cp1   = (_Float16*)carve((size_t)512 * 512 * 2);
  _Float16* w_cp2   = (_Float16*)carve((size_t)512 * 512 * 2);
  _Float16* w_cp3   = (_Float16*)carve((size_t)512 * 128 * 2);
  _Float16* w_ffn1  = (_Float16*)carve((size_t)128 * 128 * 2);
  _Float16* w_ffn2  = (_Float16*)carve((size_t)128 * 128 * 2);

  auto packb = [&](const float* srcW, _Float16* dst, int K, int N) {
    int tiles = (K >> 5) * (N >> 4);
    k_pack_b<<<(tiles + 7) / 8, 256, 0, stream>>>(srcW, dst, K, N);
  };
  packb(conv1_w, w_conv1, 128, 128);
  packb(conv2_w, w_conv2, 3200, 128);
  packb(cp_w1,   w_cp1,   512, 512);
  packb(cp_w2,   w_cp2,   512, 512);
  packb(cp_w3,   w_cp3,   512, 128);
  packb(ffn_w1,  w_ffn1,  128, 128);
  packb(ffn_w2,  w_ffn2,  128, 128);

  // feature transpose (C,H,W)->(H,W,C) per level
  const int LH[4] = {32, 16, 8, 4};
  const int LW[4] = {88, 44, 22, 11};
  const int LOFF[4] = {0, 2162688, 2703360, 2838528};
  for (int l = 0; l < 4; ++l) {
    int tot = NCAM * DD * LH[l] * LW[l];
    k_transpose_feat<<<(tot + 255) / 256, 256, 0, stream>>>(feat[l], featT + LOFF[l],
                                                            LH[l], LW[l]);
  }

  // position embedding (identical each layer) and q init
  k_pos_embed<<<QNUM, 256, 0, stream>>>(bev_pos, pe_w1, pe_b1, pe_w2, pe_b2, pe);
  k_copy<<<(QNUM * DD + 255) / 256, 256, 0, stream>>>(bev_query, q, QNUM * DD);

  // pack A then run packed GEMM (16x64 C-strip per wave)
  auto gemm = [&](const float* A, const _Float16* Bp, const float* bias, float* out_,
                  int M, int K, int N, int act) {
    int atiles = (M >> 4) * (K >> 5);
    k_pack_a<<<(atiles + 7) / 8, 256, 0, stream>>>(A, apack, M, K);
    int waves = (M >> 4) * (N >> 6);
    k_gemm_pk<<<(waves + 3) / 4, 128, 0, stream>>>(apack, Bp, bias, out_, M, K, N, act);
  };

  for (int layer = 0; layer < 2; ++layer) {
    // in_conv: qe = q + pos_embed; h = gelu(qe@conv1_w+b); h = conv5x5(h)+b
    k_add<<<(QNUM * DD + 255) / 256, 256, 0, stream>>>(q, pe, qe, QNUM * DD);
    gemm(qe, w_conv1, conv1_b, hbuf, QNUM, DD, DD, /*gelu*/2);
    {
      int waves = QNUM >> 4;            // one 16x128 strip per wave
      k_conv5x5_wmma<<<(waves + 3) / 4, 128, 0, stream>>>(hbuf, w_conv2, conv2_b, hout);
    }
    k_addnorm<<<QNUM, DD, 0, stream>>>(q, hout, n1_g, n1_b);

    // sampling heads + deformable multi-cam sampling
    k_offsw<<<QNUM, DD, 0, stream>>>(q, off_w, off_b, sw_w, sw_b, offb, swb);
    k_sample<<<QNUM, DD, 0, stream>>>(bev_pos, offb, swb, l2i, featT, accb);

    // compressor MLP: 512 -> 512 -> 512 -> 128
    gemm(accb, w_cp1, cp_b1, h1b,  QNUM, 512, 512, /*relu*/1);
    gemm(h1b,  w_cp2, cp_b2, accb, QNUM, 512, 512, /*relu*/1);
    gemm(accb, w_cp3, cp_b3, hout, QNUM, 512, DD,  /*none*/0);
    k_addnorm<<<QNUM, DD, 0, stream>>>(q, hout, n2_g, n2_b);

    // FFN
    gemm(q,    w_ffn1, ffn_b1, hbuf, QNUM, DD, DD, /*relu*/1);
    gemm(hbuf, w_ffn2, ffn_b2, hout, QNUM, DD, DD, /*none*/0);
    k_addnorm<<<QNUM, DD, 0, stream>>>(q, hout, n3_g, n3_b);
  }
}
